// SwinTransformerBlock_53841710023220
// MI455X (gfx1250) — matrix operand order repaired and checked by reading, not compiled, not run on hardware
//
#include <hip/hip_runtime.h>
#include <hip/hip_bf16.h>

// ---------------------------------------------------------------------------
// Swin transformer block for gfx1250 (MI455X): bf16 WMMA GEMMs + fused glue.
// GEMM staging uses CDNA5 async global->LDS copies (ASYNCcnt).
// ---------------------------------------------------------------------------

typedef __bf16 bf16;
typedef __attribute__((ext_vector_type(16))) __bf16 bf16x16;
typedef __attribute__((ext_vector_type(8)))  __bf16 bf16x8;
typedef __attribute__((ext_vector_type(4)))  __bf16 bf16x4;
typedef __attribute__((ext_vector_type(8)))  float  f32x8;
typedef int i32x4v __attribute__((vector_size(16)));  // builtin's pointee type

#define GLOBAL_AS __attribute__((address_space(1)))
#define LDS_AS    __attribute__((address_space(3)))

#if __has_builtin(__builtin_amdgcn_global_load_async_to_lds_b128) && \
    __has_builtin(__builtin_amdgcn_s_wait_asynccnt)
#define HAVE_ASYNC_LDS 1
#endif

__device__ __forceinline__ bf16x16 join8(bf16x8 lo, bf16x8 hi) {
  return __builtin_shufflevector(lo, hi, 0, 1, 2, 3, 4, 5, 6, 7,
                                 8, 9, 10, 11, 12, 13, 14, 15);
}

// A-fragment (16-bit, 16x32, ISA 7.12.2): lane m=L%16, hi=L/16 holds
// k = {hi*8..hi*8+7} ++ {16+hi*8..16+hi*8+7}  -> two contiguous 16B runs.
__device__ __forceinline__ bf16x16 fragA(const bf16* row, int hi) {
  bf16x8 lo = *reinterpret_cast<const bf16x8*>(row + hi * 8);
  bf16x8 hi8 = *reinterpret_cast<const bf16x8*>(row + 16 + hi * 8);
  return join8(lo, hi8);
}
// B-fragment (32x16): lane n=L%16, hi=L/16 holds
// k = {hi*8..hi*8+7} ++ {16+hi*8..16+hi*8+7}  -> two contiguous 16B runs.
__device__ __forceinline__ bf16x16 fragB(const bf16* row, int hi) {
  bf16x8 lo = *reinterpret_cast<const bf16x8*>(row + hi * 8);
  bf16x8 hi8 = *reinterpret_cast<const bf16x8*>(row + 16 + hi * 8);
  return join8(lo, hi8);
}

__device__ __forceinline__ f32x8 wmma_bf16(bf16x16 a, bf16x16 b, f32x8 c) {
  return __builtin_amdgcn_wmma_f32_16x16x32_bf16(
      /*neg_a=*/false, a, /*neg_b=*/false, b,
      /*c_mod=*/(short)0, c, /*reuse_a=*/false, /*reuse_b=*/false);
}

// 16B async global->LDS copy (CDNA5 GLOBAL_LOAD_ASYNC_TO_LDS_B128).
__device__ __forceinline__ void async_copy16(const bf16* gsrc, bf16* ldst) {
#if defined(HAVE_ASYNC_LDS)
  __builtin_amdgcn_global_load_async_to_lds_b128(
      (GLOBAL_AS i32x4v*)gsrc, (LDS_AS i32x4v*)ldst, 0, 0);
#else
  *reinterpret_cast<bf16x8*>(ldst) = *reinterpret_cast<const bf16x8*>(gsrc);
#endif
}
__device__ __forceinline__ void async_copy_wait() {
#if defined(HAVE_ASYNC_LDS)
  __builtin_amdgcn_s_wait_asynccnt(0);
#endif
}

// ---------------------------------------------------------------------------
// Kernel: fp32 W[K,N] -> bf16 Wt[N,K]  (transposed for B-fragment loads)
// ---------------------------------------------------------------------------
__global__ void wcvt_t_kernel(const float* __restrict__ in,
                              bf16* __restrict__ out, int K, int N) {
  int i = blockIdx.x * blockDim.x + threadIdx.x;  // output-ordered (coalesced)
  if (i < K * N) {
    int n = i / K, k = i - n * K;
    out[i] = (bf16)in[k * N + n];
  }
}

// ---------------------------------------------------------------------------
// Kernel: LN1 + roll(-4,-4) + window partition -> bf16 [4096*64, 128]
// one wave (32 lanes) per token, 4 channels per lane
// ---------------------------------------------------------------------------
__global__ __launch_bounds__(256)
void ln1_window_kernel(const float* __restrict__ x, const float* __restrict__ g,
                       const float* __restrict__ b, bf16* __restrict__ win) {
  int wavei = blockIdx.x * 8 + (threadIdx.x >> 5);
  int lane  = threadIdx.x & 31;
  int bimg = wavei >> 12;
  int rem  = wavei & 4095;
  int hr = rem >> 6, wr = rem & 63;            // rolled (destination) coords
  int hs = (hr + 4) & 63, ws2 = (wr + 4) & 63; // source coords in x
  const float4 v = *reinterpret_cast<const float4*>(
      &x[(((size_t)bimg * 64 + hs) * 64 + ws2) * 128 + lane * 4]);
  float s  = v.x + v.y + v.z + v.w;
  float sq = v.x * v.x + v.y * v.y + v.z * v.z + v.w * v.w;
  for (int m = 16; m >= 1; m >>= 1) {
    s  += __shfl_xor(s,  m, 32);
    sq += __shfl_xor(sq, m, 32);
  }
  float mean = s * (1.0f / 128.0f);
  float var  = sq * (1.0f / 128.0f) - mean * mean;
  float inv  = rsqrtf(var + 1e-5f);
  const float4 gg = *reinterpret_cast<const float4*>(&g[lane * 4]);
  const float4 bb = *reinterpret_cast<const float4*>(&b[lane * 4]);
  bf16x4 o;
  o[0] = (bf16)((v.x - mean) * inv * gg.x + bb.x);
  o[1] = (bf16)((v.y - mean) * inv * gg.y + bb.y);
  o[2] = (bf16)((v.z - mean) * inv * gg.z + bb.z);
  o[3] = (bf16)((v.w - mean) * inv * gg.w + bb.w);
  int widx = (bimg << 6) | ((hr >> 3) << 3) | (wr >> 3);
  int ntok = ((hr & 7) << 3) | (wr & 7);
  *reinterpret_cast<bf16x4*>(&win[((size_t)widx * 64 + ntok) * 128 + lane * 4]) = o;
}

// ---------------------------------------------------------------------------
// Kernel: LN2 -> bf16 (row-major, no remap)
// ---------------------------------------------------------------------------
__global__ __launch_bounds__(256)
void ln2_kernel(const float* __restrict__ x1, const float* __restrict__ g,
                const float* __restrict__ b, bf16* __restrict__ h2) {
  int row  = blockIdx.x * 8 + (threadIdx.x >> 5);
  int lane = threadIdx.x & 31;
  const float4 v = *reinterpret_cast<const float4*>(&x1[(size_t)row * 128 + lane * 4]);
  float s  = v.x + v.y + v.z + v.w;
  float sq = v.x * v.x + v.y * v.y + v.z * v.z + v.w * v.w;
  for (int m = 16; m >= 1; m >>= 1) {
    s  += __shfl_xor(s,  m, 32);
    sq += __shfl_xor(sq, m, 32);
  }
  float mean = s * (1.0f / 128.0f);
  float var  = sq * (1.0f / 128.0f) - mean * mean;
  float inv  = rsqrtf(var + 1e-5f);
  const float4 gg = *reinterpret_cast<const float4*>(&g[lane * 4]);
  const float4 bb = *reinterpret_cast<const float4*>(&b[lane * 4]);
  bf16x4 o;
  o[0] = (bf16)((v.x - mean) * inv * gg.x + bb.x);
  o[1] = (bf16)((v.y - mean) * inv * gg.y + bb.y);
  o[2] = (bf16)((v.z - mean) * inv * gg.z + bb.z);
  o[3] = (bf16)((v.w - mean) * inv * gg.w + bb.w);
  *reinterpret_cast<bf16x4*>(&h2[(size_t)row * 128 + lane * 4]) = o;
}

// ---------------------------------------------------------------------------
// Generic WMMA GEMM: out = epi(A[M,K](bf16) @ W(Wt[N,K] bf16) + bias)
// block = 256 threads (8 waves), tile 64(M) x 128(N), K-step 64.
// wave (wm,wn) computes 32x32 = 2x2 WMMA tiles; all frag loads are b128.
// Staging via async global->LDS copies (ASYNCcnt).
// EPI: 0 = bf16 out; 1 = proj (window-reverse + roll + residual, f32 out);
//      2 = GELU -> bf16; 3 = f32 out + residual.
// ---------------------------------------------------------------------------
template <int EPI>
__global__ __launch_bounds__(256)
void gemm_kernel(const bf16* __restrict__ A, const bf16* __restrict__ Wt,
                 const float* __restrict__ bias, void* __restrict__ outp,
                 const float* __restrict__ resid, int M, int N, int K) {
  __shared__ bf16 As[64][72];   // 72*2 = 144 B row stride (16B multiple)
  __shared__ bf16 Bt[128][72];  // transposed weight tile: [n][k]
  const int tid  = threadIdx.x;
  const int wave = tid >> 5;
  const int lane = tid & 31;
  const int wm = wave >> 2;   // 0..1
  const int wn = wave & 3;    // 0..3
  const int rowBase = blockIdx.y * 64;
  const int colBase = blockIdx.x * 128;
  const int hi = lane >> 4;
  const int nn = lane & 15;

  f32x8 c00 = {}, c01 = {}, c10 = {}, c11 = {};

  const int ar = tid >> 2, acs = (tid & 3) * 16;   // A staging: 64 rows x 64 k
  const int br = tid >> 1, bcs = (tid & 1) * 32;   // B staging: 128 n x 64 k

  for (int k0 = 0; k0 < K; k0 += 64) {
    {  // stage A tile 64x64: two 16B chunks / thread
      const bf16* src = &A[(size_t)(rowBase + ar) * K + k0 + acs];
      async_copy16(src,     &As[ar][acs]);
      async_copy16(src + 8, &As[ar][acs + 8]);
    }
    {  // stage Wt tile 128x64: four 16B chunks / thread
      const bf16* src = &Wt[(size_t)(colBase + br) * K + k0 + bcs];
      async_copy16(src,      &Bt[br][bcs]);
      async_copy16(src + 8,  &Bt[br][bcs + 8]);
      async_copy16(src + 16, &Bt[br][bcs + 16]);
      async_copy16(src + 24, &Bt[br][bcs + 24]);
    }
    if (k0 + 64 < K) {  // prefetch next K tiles (global_prefetch_b8 path)
      __builtin_prefetch(&A[(size_t)(rowBase + ar) * K + k0 + 64 + acs], 0, 1);
      __builtin_prefetch(&Wt[(size_t)(colBase + br) * K + k0 + 64 + bcs], 0, 1);
    }
    async_copy_wait();
    __syncthreads();
#pragma unroll
    for (int kk = 0; kk < 2; ++kk) {
      bf16x16 a0 = fragA(&As[wm * 32 + nn][kk * 32], hi);
      bf16x16 a1 = fragA(&As[wm * 32 + 16 + nn][kk * 32], hi);
      bf16x16 b0 = fragB(&Bt[wn * 32 + nn][kk * 32], hi);
      bf16x16 b1 = fragB(&Bt[wn * 32 + 16 + nn][kk * 32], hi);
      c00 = wmma_bf16(a0, b0, c00);
      c01 = wmma_bf16(a0, b1, c01);
      c10 = wmma_bf16(a1, b0, c10);
      c11 = wmma_bf16(a1, b1, c11);
    }
    __syncthreads();
  }

  f32x8 acc[2][2] = {{c00, c01}, {c10, c11}};
#pragma unroll
  for (int ti = 0; ti < 2; ++ti) {
#pragma unroll
    for (int tj = 0; tj < 2; ++tj) {
#pragma unroll
      for (int r = 0; r < 8; ++r) {
        int row = rowBase + wm * 32 + ti * 16 + hi * 8 + r;
        int col = colBase + wn * 32 + tj * 16 + nn;
        float v = acc[ti][tj][r] + bias[col];
        if (EPI == 0) {
          ((bf16*)outp)[(size_t)row * N + col] = (bf16)v;
        } else if (EPI == 2) {
          float u = 0.7978845608028654f * (v + 0.044715f * v * v * v);
          float gl = 0.5f * v * (1.0f + tanhf(u));
          ((bf16*)outp)[(size_t)row * N + col] = (bf16)gl;
        } else if (EPI == 3) {
          size_t idx = (size_t)row * N + col;
          ((float*)outp)[idx] = v + resid[idx];
        } else {  // EPI == 1: window-reverse + roll(+4,+4) + residual(x)
          int bimg = row >> 12;
          int widx = (row >> 6) & 63;
          int ntok = row & 63;
          int hr = ((widx >> 3) << 3) | (ntok >> 3);
          int wr = ((widx & 7) << 3) | (ntok & 7);
          int hh = (hr + 4) & 63;
          int ww = (wr + 4) & 63;
          size_t idx = (((size_t)bimg * 64 + hh) * 64 + ww) * 128 + col;
          ((float*)outp)[idx] = v + resid[idx];
        }
      }
    }
  }
}

// ---------------------------------------------------------------------------
// Kernel: shifted-window attention. 1 block / window (4096), 1 wave / head.
// QK^T: 16 WMMAs (K=32=head_dim), operands straight from global (b128).
// Softmax in registers using C/D lane layout; bias from LDS rpb; analytic
// shift mask. V is transposed into LDS once; P@V: 16 WMMAs, all-b128 frags.
// ---------------------------------------------------------------------------
__device__ __forceinline__ int swin_region(int gcoord) {
  return (gcoord < 56) ? 0 : ((gcoord < 60) ? 1 : 2);
}

__global__ __launch_bounds__(128)
void attn_kernel(const bf16* __restrict__ qkv, const float* __restrict__ rpb,
                 bf16* __restrict__ out) {
  __shared__ bf16  P[4][64][72];    // softmax probs (A-operand of P@V)
  __shared__ bf16  Vt[128][72];     // V transposed: [channel][token]
  __shared__ float rpb_s[225 * 4];
  const int tid  = threadIdx.x;
  const int head = tid >> 5;
  const int lane = tid & 31;
  const int nn = lane & 15;
  const int hi = lane >> 4;
  const size_t rowBase = (size_t)blockIdx.x * 64;
  const int widx = blockIdx.x & 63;
  const int wh = widx >> 3, ww = widx & 7;

  for (int i = tid; i < 900; i += 128) rpb_s[i] = rpb[i];

  {  // stage V transposed: thread t covers token r, 64 channels
    int r = tid >> 1, half = tid & 1;
    const bf16* src = &qkv[(rowBase + r) * 384 + 256 + half * 64];
#pragma unroll
    for (int c = 0; c < 8; ++c) {
      bf16x8 v = *reinterpret_cast<const bf16x8*>(src + c * 8);
#pragma unroll
      for (int j = 0; j < 8; ++j) Vt[half * 64 + c * 8 + j][r] = v[j];
    }
  }
  __syncthreads();

  const int qcol = head * 32;
  const int kcol = 128 + head * 32;

  // ---- S = Q K^T : 4x4 tiles of 16x16, one WMMA each (K=32) ----
  bf16x16 bK[4];
#pragma unroll
  for (int tj = 0; tj < 4; ++tj) {  // B[k][n] = K[tj*16+n][k], k contiguous
    bK[tj] = fragB(&qkv[(rowBase + tj * 16 + nn) * 384 + kcol], hi);
  }
  f32x8 s[4][4];
#pragma unroll
  for (int ti = 0; ti < 4; ++ti) {
    bf16x16 a = fragA(&qkv[(rowBase + ti * 16 + nn) * 384 + qcol], hi);
#pragma unroll
    for (int tj = 0; tj < 4; ++tj) {
      f32x8 z = {};
      s[ti][tj] = wmma_bf16(a, bK[tj], z);
    }
  }

  // ---- softmax over rows (64 logits spread over 16 lanes x 4 tiles) ----
  const float scale = 0.17677669529663687f;  // 1/sqrt(32)
#pragma unroll
  for (int ti = 0; ti < 4; ++ti) {
#pragma unroll
    for (int r = 0; r < 8; ++r) {
      int m = ti * 16 + hi * 8 + r;  // row (query token) in window
      int ihi = m >> 3, iwi = m & 7;
      int gi = swin_region(wh * 8 + ihi) * 3 + swin_region(ww * 8 + iwi);
      float l[4];
#pragma unroll
      for (int tj = 0; tj < 4; ++tj) {
        int cn = tj * 16 + nn;  // col (key token) in window
        int ihj = cn >> 3, iwj = cn & 7;
        int rel = (ihi - ihj + 7) * 15 + (iwi - iwj + 7);
        int gj = swin_region(wh * 8 + ihj) * 3 + swin_region(ww * 8 + iwj);
        l[tj] = s[ti][tj][r] * scale + rpb_s[rel * 4 + head] +
                ((gi != gj) ? -1e9f : 0.0f);
      }
      float mx = fmaxf(fmaxf(l[0], l[1]), fmaxf(l[2], l[3]));
      for (int d = 8; d >= 1; d >>= 1) mx = fmaxf(mx, __shfl_xor(mx, d, 32));
      float sum = 0.0f;
#pragma unroll
      for (int tj = 0; tj < 4; ++tj) {
        l[tj] = __expf(l[tj] - mx);
        sum += l[tj];
      }
      for (int d = 8; d >= 1; d >>= 1) sum += __shfl_xor(sum, d, 32);
      float inv = 1.0f / sum;
#pragma unroll
      for (int tj = 0; tj < 4; ++tj)
        P[head][m][tj * 16 + nn] = (bf16)(l[tj] * inv);
    }
  }
  __syncthreads();

  // ---- O = P @ V : 4x2 output tiles, K=64 -> 2 WMMAs each ----
  bf16x16 bV[2][2];
#pragma unroll
  for (int kk = 0; kk < 2; ++kk) {
#pragma unroll
    for (int tj = 0; tj < 2; ++tj) {  // B[k][n] = Vt[col][k], k contiguous
      bV[kk][tj] = fragB(&Vt[head * 32 + tj * 16 + nn][kk * 32], hi);
    }
  }
#pragma unroll
  for (int ti = 0; ti < 4; ++ti) {
    bf16x16 a0 = fragA(&P[head][ti * 16 + nn][0], hi);
    bf16x16 a1 = fragA(&P[head][ti * 16 + nn][32], hi);
#pragma unroll
    for (int tj = 0; tj < 2; ++tj) {
      f32x8 c = {};
      c = wmma_bf16(a0, bV[0][tj], c);
      c = wmma_bf16(a1, bV[1][tj], c);
#pragma unroll
      for (int r = 0; r < 8; ++r) {
        out[(rowBase + ti * 16 + hi * 8 + r) * 128 + head * 32 + tj * 16 + nn] =
            (bf16)c[r];
      }
    }
  }
}

// ---------------------------------------------------------------------------
// Workspace layout (bytes). FFN buffer aliases dead win+qkv regions.
// ---------------------------------------------------------------------------
#define OFF_WQKV  ((size_t)0)                        //  98304 B (transposed)
#define OFF_WPROJ ((size_t)98304)                    //  32768 B (transposed)
#define OFF_WFC1  ((size_t)131072)                   // 131072 B (transposed)
#define OFF_WFC2  ((size_t)262144)                   // 131072 B (transposed)
#define OFF_WIN   ((size_t)393216)                   //  64 MB (aliased by FF)
#define OFF_QKV   ((size_t)67502080)                 // 192 MB (aliased by FF)
#define OFF_FF    OFF_WIN                            // 256 MB
#define OFF_ATTN  ((size_t)268828672)                //  64 MB
#define OFF_X1    ((size_t)335937536)                // 128 MB
#define OFF_H2    ((size_t)470155264)                //  64 MB
#define WS_NEEDED ((size_t)537264128)

extern "C" void kernel_launch(void* const* d_in, const int* in_sizes, int n_in,
                              void* d_out, int out_size, void* d_ws, size_t ws_size,
                              hipStream_t stream) {
  const float* x      = (const float*)d_in[0];
  const float* ln1_g  = (const float*)d_in[1];
  const float* ln1_b  = (const float*)d_in[2];
  const float* qkv_w  = (const float*)d_in[3];
  const float* qkv_b  = (const float*)d_in[4];
  const float* proj_w = (const float*)d_in[5];
  const float* proj_b = (const float*)d_in[6];
  const float* rpb    = (const float*)d_in[7];
  const float* ln2_g  = (const float*)d_in[8];
  const float* ln2_b  = (const float*)d_in[9];
  const float* fc1_w  = (const float*)d_in[10];
  const float* fc1_b  = (const float*)d_in[11];
  const float* fc2_w  = (const float*)d_in[12];
  const float* fc2_b  = (const float*)d_in[13];

  if (ws_size < WS_NEEDED) return;
  char* ws = (char*)d_ws;
  bf16*  wqkv  = (bf16*)(ws + OFF_WQKV);
  bf16*  wproj = (bf16*)(ws + OFF_WPROJ);
  bf16*  wfc1  = (bf16*)(ws + OFF_WFC1);
  bf16*  wfc2  = (bf16*)(ws + OFF_WFC2);
  bf16*  win   = (bf16*)(ws + OFF_WIN);
  bf16*  qkvb  = (bf16*)(ws + OFF_QKV);
  bf16*  ff    = (bf16*)(ws + OFF_FF);
  bf16*  attn  = (bf16*)(ws + OFF_ATTN);
  float* x1    = (float*)(ws + OFF_X1);
  bf16*  h2    = (bf16*)(ws + OFF_H2);

  // weights -> bf16, transposed to [N][K]
  wcvt_t_kernel<<<192, 256, 0, stream>>>(qkv_w, wqkv, 128, 384);
  wcvt_t_kernel<<<64,  256, 0, stream>>>(proj_w, wproj, 128, 128);
  wcvt_t_kernel<<<256, 256, 0, stream>>>(fc1_w, wfc1, 128, 512);
  wcvt_t_kernel<<<256, 256, 0, stream>>>(fc2_w, wfc2, 512, 128);

  // LN1 + roll(-4,-4) + window partition
  ln1_window_kernel<<<32768, 256, 0, stream>>>(x, ln1_g, ln1_b, win);

  // QKV: [262144,128] @ [128,384]
  gemm_kernel<0><<<dim3(3, 4096), 256, 0, stream>>>(
      win, wqkv, qkv_b, (void*)qkvb, nullptr, 262144, 384, 128);

  // shifted-window attention (4096 windows, 4 heads)
  attn_kernel<<<4096, 128, 0, stream>>>(qkvb, rpb, attn);

  // proj + window reverse + roll(+4,+4) + residual(x) -> x1
  gemm_kernel<1><<<dim3(1, 4096), 256, 0, stream>>>(
      attn, wproj, proj_b, (void*)x1, x, 262144, 128, 128);

  // LN2
  ln2_kernel<<<32768, 256, 0, stream>>>(x1, ln2_g, ln2_b, h2);

  // fc1 + GELU: [262144,128] @ [128,512]
  gemm_kernel<2><<<dim3(4, 4096), 256, 0, stream>>>(
      h2, wfc1, fc1_b, (void*)ff, nullptr, 262144, 512, 128);

  // fc2 + residual(x1): [262144,512] @ [512,128] -> d_out (fp32)
  gemm_kernel<3><<<dim3(1, 4096), 256, 0, stream>>>(
      ff, wfc2, fc2_b, d_out, x1, 262144, 128, 512);
}
